// ProbabilisticBridgeTrainGNN_86603720556694
// MI455X (gfx1250) — matrix operand order repaired and checked by reading, not compiled, not run on hardware
//
#include <hip/hip_runtime.h>
#include <math.h>

// ---------------------------------------------------------------------------
// ProbabilisticBridgeTrainGNN forward for MI455X (gfx1250, wave32, WMMA).
// Dense GEMMs: v_wmma_f32_16x16x32_f16, weights pre-staged to half in
// [col][k] layout (contiguous 32B B-fragments), activations fp32 loaded as
// float4 and packed-converted. Branchless inner loops, 4 col-tiles per wave.
// LSTM recurrence: persistent workgroup, Whh fragments hoisted into VGPRs.
// ---------------------------------------------------------------------------

typedef __attribute__((ext_vector_type(16))) _Float16 v16h;
typedef __attribute__((ext_vector_type(8)))  float    v8f;

#define NNODES 16384
#define NFEAT  64
#define TFEAT  16
#define HDIM   128
#define NEDGE  131072
#define NHEADS 8

__device__ __forceinline__ float sigf(float x) { return 1.0f / (1.0f + expf(-x)); }

__device__ __forceinline__ float act_apply(float v, int act) {
  if (act == 1) return fmaxf(v, 0.0f);                                      // relu
  if (act == 2) return v > 0.0f ? v : 0.1f * v;                             // leaky 0.1
  if (act == 3) return 0.5f * v * (1.0f + erff(v * 0.70710678118654752f));  // exact gelu
  return v;
}

// ---------------------------------------------------------------------------
// Weight staging
// ---------------------------------------------------------------------------
// out[n*Kpad + k] = (k < Kin) ? (half)in[k*N + n] : 0   (in: Kin x N row-major)
__global__ void wt_conv(const float* __restrict__ in, _Float16* __restrict__ out,
                        int N, int Kin, int Kpad) {
  int idx = blockIdx.x * 256 + threadIdx.x;
  if (idx >= N * Kpad) return;
  int n = idx / Kpad, k = idx - n * Kpad;
  out[idx] = (k < Kin) ? (_Float16)in[(size_t)k * N + n] : (_Float16)0.0f;
}

// plain convert (weights already [out][in] row-major: Wih, Whh)
__global__ void h_conv(const float* __restrict__ in, _Float16* __restrict__ out, int n) {
  int i = blockIdx.x * 256 + threadIdx.x;
  if (i < n) out[i] = (_Float16)in[i];
}

// corr[n] = sum_k gatb[k] * Wp[k*128+n]   (folds GAT output bias through proj)
__global__ void bias_corr(const float* __restrict__ gatb, const float* __restrict__ wp,
                          float* __restrict__ corr) {
  int n = threadIdx.x;  // 128
  float s = 0.0f;
  for (int k = 0; k < NHEADS * HDIM; ++k) s += gatb[k] * wp[(size_t)k * HDIM + n];
  corr[n] = s;
}

// tfp[n*32+j] = j<16 ? x[n*80+64+j] : 0   (zero-padded train features, K=32)
__global__ void tfp_kernel(const float* __restrict__ x, float* __restrict__ tfp) {
  int idx = blockIdx.x * 256 + threadIdx.x;  // N*32
  int n = idx >> 5, j = idx & 31;
  tfp[idx] = (j < TFEAT) ? x[(size_t)n * (NFEAT + TFEAT) + NFEAT + j] : 0.0f;
}

// ---------------------------------------------------------------------------
// WMMA GEMM: C[M x Nc] = act( A @ Bh + bias + bias2 )
// A fp32 row-major (lda), K multiple of 32 (branchless).
// Bh half in [col][K] layout. One wave per 16-row x 64-col macro tile
// (4 col tiles share one A fragment). grid = (Nc/64, M/16), block = 32.
// ---------------------------------------------------------------------------
__global__ void wmma_gemm(const float* __restrict__ A, int lda,
                          const _Float16* __restrict__ Bh,
                          const float* __restrict__ bias,
                          const float* __restrict__ bias2,
                          float* __restrict__ C, int ldc, int K, int act) {
  const int lane = threadIdx.x & 31;
  const int hi   = lane >> 4;
  const int l15  = lane & 15;
  const int m0   = blockIdx.y << 4;
  const int n0   = blockIdx.x << 6;

  const float* Ap = A + (size_t)(m0 + l15) * lda + (hi << 4);
  const _Float16* Bp0 = Bh + (size_t)(n0 +  0 + l15) * K + (hi << 4);
  const _Float16* Bp1 = Bh + (size_t)(n0 + 16 + l15) * K + (hi << 4);
  const _Float16* Bp2 = Bh + (size_t)(n0 + 32 + l15) * K + (hi << 4);
  const _Float16* Bp3 = Bh + (size_t)(n0 + 48 + l15) * K + (hi << 4);

  v8f acc0 = {}, acc1 = {}, acc2 = {}, acc3 = {};
  for (int k0 = 0; k0 < K; k0 += 32) {
    float4 a0 = *(const float4*)(Ap + k0);
    float4 a1 = *(const float4*)(Ap + k0 + 4);
    float4 a2 = *(const float4*)(Ap + k0 + 8);
    float4 a3 = *(const float4*)(Ap + k0 + 12);
    v16h a;
    a[0]  = (_Float16)a0.x; a[1]  = (_Float16)a0.y; a[2]  = (_Float16)a0.z; a[3]  = (_Float16)a0.w;
    a[4]  = (_Float16)a1.x; a[5]  = (_Float16)a1.y; a[6]  = (_Float16)a1.z; a[7]  = (_Float16)a1.w;
    a[8]  = (_Float16)a2.x; a[9]  = (_Float16)a2.y; a[10] = (_Float16)a2.z; a[11] = (_Float16)a2.w;
    a[12] = (_Float16)a3.x; a[13] = (_Float16)a3.y; a[14] = (_Float16)a3.z; a[15] = (_Float16)a3.w;
    v16h b0 = *(const v16h*)(Bp0 + k0);
    v16h b1 = *(const v16h*)(Bp1 + k0);
    v16h b2 = *(const v16h*)(Bp2 + k0);
    v16h b3 = *(const v16h*)(Bp3 + k0);
    acc0 = __builtin_amdgcn_wmma_f32_16x16x32_f16(false, a, false, b0, (short)0, acc0, false, false);
    acc1 = __builtin_amdgcn_wmma_f32_16x16x32_f16(false, a, false, b1, (short)0, acc1, false, false);
    acc2 = __builtin_amdgcn_wmma_f32_16x16x32_f16(false, a, false, b2, (short)0, acc2, false, false);
    acc3 = __builtin_amdgcn_wmma_f32_16x16x32_f16(false, a, false, b3, (short)0, acc3, false, false);
  }

#pragma unroll
  for (int tt = 0; tt < 4; ++tt) {
    const v8f acc = tt == 0 ? acc0 : (tt == 1 ? acc1 : (tt == 2 ? acc2 : acc3));
    int col = n0 + tt * 16 + l15;
    float bv = 0.0f;
    if (bias)  bv += bias[col];
    if (bias2) bv += bias2[col];
#pragma unroll
    for (int r = 0; r < 8; ++r) {
      int row = m0 + r + (hi << 3);
      C[(size_t)row * ldc + col] = act_apply(acc[r] + bv, act);
    }
  }
}

// ---------------------------------------------------------------------------
// Elementwise / graph kernels
// ---------------------------------------------------------------------------
__global__ void fill_kernel(float* p, float v, int n) {
  int i = blockIdx.x * 256 + threadIdx.x;
  if (i < n) p[i] = v;
}

__global__ void deg_kernel(const int* __restrict__ dst, float* deg, int E) {
  int i = blockIdx.x * 256 + threadIdx.x;
  if (i < E) atomicAdd(&deg[dst[i]], 1.0f);
}

__global__ void rsqrt_kernel(float* p, int n) {
  int i = blockIdx.x * 256 + threadIdx.x;
  if (i < n) p[i] = rsqrtf(p[i]);
}

__global__ void gcn_agg(const int* __restrict__ src, const int* __restrict__ dst,
                        int E, const float* __restrict__ dis,
                        const float* __restrict__ t, float* __restrict__ out) {
  int row = blockIdx.x;
  int j = threadIdx.x;  // 128
  int s, d;
  if (row < E) { s = src[row]; d = dst[row]; } else { s = d = row - E; }
  float norm = dis[s] * dis[d];
  atomicAdd(&out[(size_t)d * HDIM + j], t[(size_t)s * HDIM + j] * norm);
}

__global__ void bn_relu(const float* __restrict__ agg, const float* __restrict__ convb,
                        const float* __restrict__ g, const float* __restrict__ b,
                        const float* __restrict__ rm, const float* __restrict__ rv,
                        const float* __restrict__ res, float* __restrict__ out,
                        float* __restrict__ resout) {
  size_t i = (size_t)blockIdx.x * 256 + threadIdx.x;
  int j = (int)(i & (HDIM - 1));
  float v = agg[i] + convb[j];
  v = (v - rm[j]) * g[j] * rsqrtf(rv[j] + 1e-5f) + b[j];
  if (res) v += res[i];
  v = fmaxf(v, 0.0f);
  out[i] = v;
  if (resout) resout[i] = v;
}

__global__ void gat_att(const float* __restrict__ gath,
                        const float* __restrict__ att_src,
                        const float* __restrict__ att_dst,
                        float* __restrict__ asrc, float* __restrict__ adst) {
  int idx = blockIdx.x * 256 + threadIdx.x;  // N*8
  int n = idx >> 3, hd = idx & 7;
  const float* hv = gath + (size_t)n * (NHEADS * HDIM) + hd * HDIM;
  const float* ws = att_src + hd * HDIM;
  const float* wd = att_dst + hd * HDIM;
  float ss = 0.0f, sd = 0.0f;
  for (int j = 0; j < HDIM; ++j) { float v = hv[j]; ss += v * ws[j]; sd += v * wd[j]; }
  asrc[idx] = ss; adst[idx] = sd;
}

__device__ __forceinline__ void atomicMaxF(float* addr, float val) {
  int old = __float_as_int(*addr);
  while (__int_as_float(old) < val) {
    int prev = atomicCAS((int*)addr, old, __float_as_int(val));
    if (prev == old) break;
    old = prev;
  }
}

__global__ void gat_max(const int* __restrict__ src, const int* __restrict__ dst,
                        int E, int ET, const float* __restrict__ asrc,
                        const float* __restrict__ adst, float* __restrict__ m) {
  int idx = blockIdx.x * 256 + threadIdx.x;
  if (idx >= ET * NHEADS) return;
  int e = idx >> 3, hd = idx & 7;
  int s, d;
  if (e < E) { s = src[e]; d = dst[e]; } else { s = d = e - E; }
  float ev = asrc[s * NHEADS + hd] + adst[d * NHEADS + hd];
  ev = ev > 0.0f ? ev : 0.2f * ev;  // leaky 0.2
  atomicMaxF(&m[d * NHEADS + hd], ev);
}

__global__ void gat_exp(const int* __restrict__ src, const int* __restrict__ dst,
                        int E, int ET, const float* __restrict__ asrc,
                        const float* __restrict__ adst, const float* __restrict__ m,
                        float* __restrict__ exbuf, float* __restrict__ ssum) {
  int idx = blockIdx.x * 256 + threadIdx.x;
  if (idx >= ET * NHEADS) return;
  int e = idx >> 3, hd = idx & 7;
  int s, d;
  if (e < E) { s = src[e]; d = dst[e]; } else { s = d = e - E; }
  float ev = asrc[s * NHEADS + hd] + adst[d * NHEADS + hd];
  ev = ev > 0.0f ? ev : 0.2f * ev;
  float ex = expf(ev - m[d * NHEADS + hd]);
  exbuf[idx] = ex;
  atomicAdd(&ssum[d * NHEADS + hd], ex);
}

__global__ void gat_scatter(const int* __restrict__ src, const int* __restrict__ dst,
                            int E, const float* __restrict__ gath,
                            const float* __restrict__ exbuf,
                            const float* __restrict__ ssum, float* __restrict__ out) {
  int e = blockIdx.x;
  int tid = threadIdx.x;  // 256
  int s, d;
  if (e < E) { s = src[e]; d = dst[e]; } else { s = d = e - E; }
#pragma unroll
  for (int i = 0; i < 4; ++i) {
    int v = tid + i * 256;  // 0..1023
    int hd = v >> 7;
    float alpha = exbuf[(size_t)e * NHEADS + hd] / ssum[d * NHEADS + hd];
    atomicAdd(&out[(size_t)d * 1024 + v], alpha * gath[(size_t)s * 1024 + v]);
  }
}

__global__ void seq_transpose(const float* __restrict__ in, float* __restrict__ out) {
  int n = blockIdx.x;
  int j = threadIdx.x;  // 128
  int b = n >> 10, t = n & 1023;
  out[(size_t)(t * 16 + b) * HDIM + j] = in[(size_t)n * HDIM + j];
}

__global__ void head_dot(const float* __restrict__ A, int lda, int K,
                         const float* __restrict__ w, const float* __restrict__ b,
                         float* __restrict__ out, int stride) {
  int n = blockIdx.x * 256 + threadIdx.x;
  const float* a = A + (size_t)n * lda;
  float s = 0.0f;
  for (int j = 0; j < K; ++j) s += a[j] * w[j];
  out[(size_t)n * stride] = s + b[0];
}

// ---------------------------------------------------------------------------
// Persistent LSTM recurrence: one 256-thread workgroup per (lstm,dir).
// gates = gx[t] + h @ Whh^T via WMMA; Whh (half, [gate][k]) fragments hoisted.
// ---------------------------------------------------------------------------
struct RecArgs {
  const float*    gx[4];
  const _Float16* whh[4];
  float*          out[4];
  int             rev[4];
  int             dir[4];
  int             final_layout;  // 0: out row = t*16+b ; 1: out row = b*1024+t
};

__global__ void __launch_bounds__(256) lstm_rec(RecArgs args) {
  __shared__ float hbuf[16 * 128];
  __shared__ float gbuf[16 * 512];

  const int blk = blockIdx.x;
  const float*    gx  = args.gx[blk];
  const _Float16* whh = args.whh[blk];
  float*          out = args.out[blk];
  const int rev    = args.rev[blk];
  const int dofs   = args.dir[blk] * 128;
  const int finalL = args.final_layout;

  const int tid  = threadIdx.x;
  const int wave = tid >> 5;
  const int lane = tid & 31;
  const int hi   = lane >> 4;
  const int l15  = lane & 15;

  // Hoist Whh^T fragments: wave handles n-tiles wave*4 .. wave*4+3 of 512 cols
  v16h bfrag[4][4];
#pragma unroll
  for (int tt = 0; tt < 4; ++tt) {
    int ncol = (wave * 4 + tt) * 16 + l15;
#pragma unroll
    for (int ks = 0; ks < 4; ++ks)
      bfrag[tt][ks] = *(const v16h*)(whh + (size_t)ncol * 128 + ks * 32 + (hi << 4));
  }

  for (int i = tid; i < 16 * 128; i += 256) hbuf[i] = 0.0f;
  float creg[8];
#pragma unroll
  for (int q = 0; q < 8; ++q) creg[q] = 0.0f;
  __syncthreads();

  const v8f zero8 = {};
  for (int step = 0; step < 1024; ++step) {
    const int t = rev ? (1023 - step) : step;

    v8f acc[4];
    acc[0] = zero8; acc[1] = zero8; acc[2] = zero8; acc[3] = zero8;
#pragma unroll
    for (int ks = 0; ks < 4; ++ks) {
      const float* hp = &hbuf[l15 * 128 + ks * 32 + (hi << 4)];
      float4 a0 = *(const float4*)(hp);
      float4 a1 = *(const float4*)(hp + 4);
      float4 a2 = *(const float4*)(hp + 8);
      float4 a3 = *(const float4*)(hp + 12);
      v16h a;
      a[0]  = (_Float16)a0.x; a[1]  = (_Float16)a0.y; a[2]  = (_Float16)a0.z; a[3]  = (_Float16)a0.w;
      a[4]  = (_Float16)a1.x; a[5]  = (_Float16)a1.y; a[6]  = (_Float16)a1.z; a[7]  = (_Float16)a1.w;
      a[8]  = (_Float16)a2.x; a[9]  = (_Float16)a2.y; a[10] = (_Float16)a2.z; a[11] = (_Float16)a2.w;
      a[12] = (_Float16)a3.x; a[13] = (_Float16)a3.y; a[14] = (_Float16)a3.z; a[15] = (_Float16)a3.w;
#pragma unroll
      for (int tt = 0; tt < 4; ++tt)
        acc[tt] = __builtin_amdgcn_wmma_f32_16x16x32_f16(false, a, false,
                    bfrag[tt][ks], (short)0, acc[tt], false, false);
    }

    const float* gxt = gx + (size_t)t * 16 * 512;
#pragma unroll
    for (int tt = 0; tt < 4; ++tt) {
      int col = (wave * 4 + tt) * 16 + l15;
#pragma unroll
      for (int r = 0; r < 8; ++r) {
        int row = r + (hi << 3);
        gbuf[row * 512 + col] = acc[tt][r] + gxt[row * 512 + col];
      }
    }
    __syncthreads();

#pragma unroll
    for (int q = 0; q < 8; ++q) {
      int idx = tid * 8 + q;  // 0..2047 over 16x128
      int b = idx >> 7, j = idx & 127;
      float gi = gbuf[b * 512 + j];
      float gf = gbuf[b * 512 + 128 + j];
      float gg = gbuf[b * 512 + 256 + j];
      float go = gbuf[b * 512 + 384 + j];
      float c = sigf(gf) * creg[q] + sigf(gi) * tanhf(gg);
      creg[q] = c;
      float hv = sigf(go) * tanhf(c);
      hbuf[b * 128 + j] = hv;
      size_t row = finalL ? ((size_t)b * 1024 + t) : ((size_t)t * 16 + b);
      out[row * 256 + dofs + j] = hv;
    }
    __syncthreads();
  }
}

// ---------------------------------------------------------------------------
// Launch. Input leaves flattened in setup_inputs() insertion order (89 leaves):
// 0:x 1:edge_index 2:batch, then conv1{W,b}, bn1{g,b,rm,rv}, conv2, bn2,
// conv3, bn3, gat{W,att_src,att_dst,b}, gat_proj, train_feat, fusion, shared,
// lstm_u[2 layers x (fwd,bwd) x {Wih,Whh,bih,bhh}], lstm_d[...], acc_mean,
// acc_logvar, force_mean, force_logvar, unload_mean{l1,l2}, unload_logvar,
// derail_mean, derail_logvar.
// ---------------------------------------------------------------------------
extern "C" void kernel_launch(void* const* d_in, const int* in_sizes, int n_in,
                              void* d_out, int out_size, void* d_ws, size_t ws_size,
                              hipStream_t stream) {
  (void)in_sizes; (void)n_in; (void)out_size; (void)ws_size;
#define PF(i) ((const float*)d_in[(i)])
  const float* x  = PF(0);
  const int*   ei = (const int*)d_in[1];
  const int N = NNODES, E = NEDGE, ET = NEDGE + NNODES;
  const int* src = ei;
  const int* dst = ei + E;

  // ---- workspace layout (floats) ----
  float* W = (float*)d_ws;
  float* DIS  = W;                        // N
  float* T0   = DIS + N;                  // N*128 (gemm/MLP scratch)
  float* AGG  = T0 + (size_t)N * 128;     // N*128
  float* HB   = AGG + (size_t)N * 128;    // N*128
  float* RES  = HB + (size_t)N * 128;     // N*128
  float* SHRD = RES + (size_t)N * 128;    // N*128
  float* CAT  = SHRD + (size_t)N * 128;   // N*256
  float* UF   = CAT + (size_t)N * 256;    // N*256
  float* DF   = UF + (size_t)N * 256;     // N*256
  float* BIG  = DF + (size_t)N * 256;
  // GAT phase view of BIG
  float* GATH   = BIG;                        // N*1024
  float* GATAGG = GATH + (size_t)N * 1024;    // N*1024
  float* ASRC   = GATAGG + (size_t)N * 1024;  // N*8
  float* ADST   = ASRC + (size_t)N * 8;
  float* MMAX   = ADST + (size_t)N * 8;
  float* SSUM   = MMAX + (size_t)N * 8;
  float* EXBUF  = SSUM + (size_t)N * 8;       // ET*8
  // LSTM phase view of BIG (GAT buffers dead by then)
  float* XS0  = BIG;                          // N*128
  float* XS1U = XS0 + (size_t)N * 128;        // N*256
  float* XS1D = XS1U + (size_t)N * 256;       // N*256
  float* GX[4] = { XS1D + (size_t)N * 256,
                   XS1D + (size_t)N * 256 + (size_t)N * 512,
                   XS1D + (size_t)N * 256 + (size_t)N * 1024,
                   XS1D + (size_t)N * 256 + (size_t)N * 1536 };  // N*512 each

  // ---- half-precision weight arena (after the BIG union region) ----
  _Float16* hp = (_Float16*)(BIG + (size_t)N * 2688);
  auto halloc = [&](size_t n) { _Float16* r = hp; hp += n; return r; };
  _Float16* WHc1  = halloc(128 * 64);
  _Float16* WHc2  = halloc(128 * 128);
  _Float16* WHc3  = halloc(128 * 128);
  _Float16* WHgat = halloc(1024 * 128);
  _Float16* WHgp  = halloc(128 * 1024);
  _Float16* WHtf  = halloc(128 * 32);
  _Float16* WHfu  = halloc(128 * 256);
  _Float16* WHsh  = halloc(128 * 128);
  _Float16 *WHih0[4], *WHih1[4], *WHhh0[4], *WHhh1[4], *WHml[4];
  for (int i = 0; i < 4; ++i) WHih0[i] = halloc(512 * 128);
  for (int i = 0; i < 4; ++i) WHih1[i] = halloc(512 * 256);
  for (int i = 0; i < 4; ++i) WHhh0[i] = halloc(512 * 128);
  for (int i = 0; i < 4; ++i) WHhh1[i] = halloc(512 * 128);
  for (int i = 0; i < 4; ++i) WHml[i]  = halloc(128 * 256);
  float* TFP  = (float*)hp;        // N*32 padded train features
  float* CORR = TFP + (size_t)N * 32;  // 128

  dim3 blk32(32), blk128(128), blk256(256);
  auto g1 = [](int n) { return dim3((n + 255) / 256); };

  // ---- stage weights (half, [col][k]) ----
  const int ihW0[4] = {33, 37, 49, 53}, hhW0[4] = {34, 38, 50, 54};
  const int bi0[4]  = {35, 39, 51, 55}, bh0[4]  = {36, 40, 52, 56};
  const int ihW1[4] = {41, 45, 57, 61}, hhW1[4] = {42, 46, 58, 62};
  const int bi1[4]  = {43, 47, 59, 63}, bh1[4]  = {44, 48, 60, 64};
  const int mlW[4]  = {73, 77, 81, 85};

  wt_conv<<<g1(128 * 64), blk256, 0, stream>>>(PF(3), WHc1, 128, 64, 64);
  wt_conv<<<g1(128 * 128), blk256, 0, stream>>>(PF(9), WHc2, 128, 128, 128);
  wt_conv<<<g1(128 * 128), blk256, 0, stream>>>(PF(15), WHc3, 128, 128, 128);
  wt_conv<<<g1(1024 * 128), blk256, 0, stream>>>(PF(21), WHgat, 1024, 128, 128);
  wt_conv<<<g1(128 * 1024), blk256, 0, stream>>>(PF(25), WHgp, 128, 1024, 1024);
  wt_conv<<<g1(128 * 32), blk256, 0, stream>>>(PF(27), WHtf, 128, 16, 32);
  wt_conv<<<g1(128 * 256), blk256, 0, stream>>>(PF(29), WHfu, 128, 256, 256);
  wt_conv<<<g1(128 * 128), blk256, 0, stream>>>(PF(31), WHsh, 128, 128, 128);
  for (int i = 0; i < 4; ++i) {
    h_conv<<<g1(512 * 128), blk256, 0, stream>>>(PF(ihW0[i]), WHih0[i], 512 * 128);
    h_conv<<<g1(512 * 256), blk256, 0, stream>>>(PF(ihW1[i]), WHih1[i], 512 * 256);
    h_conv<<<g1(512 * 128), blk256, 0, stream>>>(PF(hhW0[i]), WHhh0[i], 512 * 128);
    h_conv<<<g1(512 * 128), blk256, 0, stream>>>(PF(hhW1[i]), WHhh1[i], 512 * 128);
    wt_conv<<<g1(128 * 256), blk256, 0, stream>>>(PF(mlW[i]), WHml[i], 128, 256, 256);
  }
  bias_corr<<<dim3(1), blk128, 0, stream>>>(PF(24), PF(25), CORR);
  tfp_kernel<<<g1(N * 32), blk256, 0, stream>>>(x, TFP);

  // ---- degree / normalization ----
  fill_kernel<<<g1(N), blk256, 0, stream>>>(DIS, 1.0f, N);  // self loops
  deg_kernel<<<g1(E), blk256, 0, stream>>>(dst, DIS, E);
  rsqrt_kernel<<<g1(N), blk256, 0, stream>>>(DIS, N);

  // ---- GCN layer 1 ----
  wmma_gemm<<<dim3(2, 1024), blk32, 0, stream>>>(x, 80, WHc1, nullptr, nullptr, T0, 128, 64, 0);
  fill_kernel<<<g1(N * 128), blk256, 0, stream>>>(AGG, 0.0f, N * 128);
  gcn_agg<<<dim3(ET), blk128, 0, stream>>>(src, dst, E, DIS, T0, AGG);
  bn_relu<<<g1(N * 128), blk256, 0, stream>>>(AGG, PF(4), PF(5), PF(6), PF(7), PF(8),
      nullptr, HB, RES);
  // ---- GCN layer 2 ----
  wmma_gemm<<<dim3(2, 1024), blk32, 0, stream>>>(HB, 128, WHc2, nullptr, nullptr, T0, 128, 128, 0);
  fill_kernel<<<g1(N * 128), blk256, 0, stream>>>(AGG, 0.0f, N * 128);
  gcn_agg<<<dim3(ET), blk128, 0, stream>>>(src, dst, E, DIS, T0, AGG);
  bn_relu<<<g1(N * 128), blk256, 0, stream>>>(AGG, PF(10), PF(11), PF(12), PF(13), PF(14),
      nullptr, HB, nullptr);
  // ---- GCN layer 3 (+ residual) ----
  wmma_gemm<<<dim3(2, 1024), blk32, 0, stream>>>(HB, 128, WHc3, nullptr, nullptr, T0, 128, 128, 0);
  fill_kernel<<<g1(N * 128), blk256, 0, stream>>>(AGG, 0.0f, N * 128);
  gcn_agg<<<dim3(ET), blk128, 0, stream>>>(src, dst, E, DIS, T0, AGG);
  bn_relu<<<g1(N * 128), blk256, 0, stream>>>(AGG, PF(16), PF(17), PF(18), PF(19), PF(20),
      RES, HB, nullptr);

  // ---- GAT ----
  wmma_gemm<<<dim3(16, 1024), blk32, 0, stream>>>(HB, 128, WHgat, nullptr, nullptr,
      GATH, 1024, 128, 0);
  gat_att<<<g1(N * 8), blk256, 0, stream>>>(GATH, PF(22), PF(23), ASRC, ADST);
  fill_kernel<<<g1(N * 8), blk256, 0, stream>>>(MMAX, -1e30f, N * 8);
  fill_kernel<<<g1(N * 8), blk256, 0, stream>>>(SSUM, 0.0f, N * 8);
  fill_kernel<<<g1(N * 1024), blk256, 0, stream>>>(GATAGG, 0.0f, N * 1024);
  gat_max<<<g1(ET * 8), blk256, 0, stream>>>(src, dst, E, ET, ASRC, ADST, MMAX);
  gat_exp<<<g1(ET * 8), blk256, 0, stream>>>(src, dst, E, ET, ASRC, ADST, MMAX, EXBUF, SSUM);
  gat_scatter<<<dim3(ET), blk256, 0, stream>>>(src, dst, E, GATH, EXBUF, SSUM, GATAGG);
  // agg @ Wp + proj.b + (gat.b @ Wp) -> relu -> CAT[:, 0:128]
  wmma_gemm<<<dim3(2, 1024), blk32, 0, stream>>>(GATAGG, 1024, WHgp, PF(26), CORR,
      CAT, 256, 1024, 1);
  // train_feat (zero-padded K=32) -> relu -> CAT[:, 128:256]
  wmma_gemm<<<dim3(2, 1024), blk32, 0, stream>>>(TFP, 32, WHtf, PF(28), nullptr,
      CAT + 128, 256, 32, 1);
  // fusion, shared
  wmma_gemm<<<dim3(2, 1024), blk32, 0, stream>>>(CAT, 256, WHfu, PF(30), nullptr,
      HB, 128, 256, 1);
  wmma_gemm<<<dim3(2, 1024), blk32, 0, stream>>>(HB, 128, WHsh, PF(32), nullptr,
      SHRD, 128, 128, 1);

  // ---- LSTMs ----
  seq_transpose<<<dim3(N), blk128, 0, stream>>>(SHRD, XS0);

  for (int i = 0; i < 4; ++i)
    wmma_gemm<<<dim3(8, 1024), blk32, 0, stream>>>(XS0, 128, WHih0[i],
        PF(bi0[i]), PF(bh0[i]), GX[i], 512, 128, 0);

  RecArgs r0;
  for (int i = 0; i < 4; ++i) { r0.gx[i] = GX[i]; r0.whh[i] = WHhh0[i]; }
  r0.out[0] = XS1U; r0.out[1] = XS1U; r0.out[2] = XS1D; r0.out[3] = XS1D;
  r0.rev[0] = 0; r0.rev[1] = 1; r0.rev[2] = 0; r0.rev[3] = 1;
  r0.dir[0] = 0; r0.dir[1] = 1; r0.dir[2] = 0; r0.dir[3] = 1;
  r0.final_layout = 0;
  lstm_rec<<<dim3(4), blk256, 0, stream>>>(r0);

  const float* xin1[4] = {XS1U, XS1U, XS1D, XS1D};
  for (int i = 0; i < 4; ++i)
    wmma_gemm<<<dim3(8, 1024), blk32, 0, stream>>>(xin1[i], 256, WHih1[i],
        PF(bi1[i]), PF(bh1[i]), GX[i], 512, 256, 0);

  RecArgs r1;
  for (int i = 0; i < 4; ++i) { r1.gx[i] = GX[i]; r1.whh[i] = WHhh1[i]; }
  r1.out[0] = UF; r1.out[1] = UF; r1.out[2] = DF; r1.out[3] = DF;
  r1.rev[0] = 0; r1.rev[1] = 1; r1.rev[2] = 0; r1.rev[3] = 1;
  r1.dir[0] = 0; r1.dir[1] = 1; r1.dir[2] = 0; r1.dir[3] = 1;
  r1.final_layout = 1;
  lstm_rec<<<dim3(4), blk256, 0, stream>>>(r1);

  // ---- heads ----
  float* OUT = (float*)d_out;  // pred_mean [N,4] then pred_logvar [N,4]
  float* OLV = OUT + (size_t)N * 4;
  dim3 gh(N / 256);
  head_dot<<<gh, blk256, 0, stream>>>(SHRD, 128, 128, PF(65), PF(66), OUT + 0, 4);
  head_dot<<<gh, blk256, 0, stream>>>(SHRD, 128, 128, PF(67), PF(68), OLV + 0, 4);
  head_dot<<<gh, blk256, 0, stream>>>(SHRD, 128, 128, PF(69), PF(70), OUT + 3, 4);
  head_dot<<<gh, blk256, 0, stream>>>(SHRD, 128, 128, PF(71), PF(72), OLV + 3, 4);
  // unload (col 2): MLP(uf), leaky 0.1
  wmma_gemm<<<dim3(2, 1024), blk32, 0, stream>>>(UF, 256, WHml[0], PF(74), nullptr,
      T0, 128, 256, 2);
  head_dot<<<gh, blk256, 0, stream>>>(T0, 128, 128, PF(75), PF(76), OUT + 2, 4);
  wmma_gemm<<<dim3(2, 1024), blk32, 0, stream>>>(UF, 256, WHml[1], PF(78), nullptr,
      T0, 128, 256, 2);
  head_dot<<<gh, blk256, 0, stream>>>(T0, 128, 128, PF(79), PF(80), OLV + 2, 4);
  // derail (col 1): MLP(df), exact gelu
  wmma_gemm<<<dim3(2, 1024), blk32, 0, stream>>>(DF, 256, WHml[2], PF(82), nullptr,
      T0, 128, 256, 3);
  head_dot<<<gh, blk256, 0, stream>>>(T0, 128, 128, PF(83), PF(84), OUT + 1, 4);
  wmma_gemm<<<dim3(2, 1024), blk32, 0, stream>>>(DF, 256, WHml[3], PF(86), nullptr,
      T0, 128, 256, 3);
  head_dot<<<gh, blk256, 0, stream>>>(T0, 128, 128, PF(87), PF(88), OLV + 1, 4);
#undef PF
}
